// DrugGAE_two_smile_78503412236482
// MI455X (gfx1250) — matrix operand order repaired and checked by reading, not compiled, not run on hardware
//
#include <hip/hip_runtime.h>

// ---------------- problem constants ----------------
#define G_N   3040
#define A_N   100
#define FS_N  75
#define HS_N  128

// ---------------- WMMA fragment machinery ----------------
typedef __attribute__((ext_vector_type(16))) __bf16          bf16x16;
typedef __attribute__((ext_vector_type(8)))  float           f32x8;
typedef __attribute__((ext_vector_type(8)))  unsigned short  us8;

union FragBF { bf16x16 v; us8 h[2]; unsigned short s[16]; };
union FragF  { f32x8 v; float f[8]; };

__device__ __forceinline__ unsigned short f2b(float x) {
  union { float f; unsigned u; } v; v.f = x;
  unsigned r = v.u + 0x7FFFu + ((v.u >> 16) & 1u);   // RNE truncate f32 -> bf16
  return (unsigned short)(r >> 16);
}

// A operand, 16x32 bf16 tile from row-major LDS [rows][ld] (ld = multiple of 8).
// ISA layout: lanes 0-15 M rows; VGPR0-3 hold K 0..7 (lo lanes) / 8..15 (hi lanes),
// VGPR4-7 hold K 16..23 / 24..31  -> two contiguous b128 loads per lane.
__device__ __forceinline__ void loadA_frag(const unsigned short* base, int ld,
                                           int m0, int k0, FragBF& f) {
  const int lane = threadIdx.x & 31;
  const int mr = m0 + (lane & 15);
  const int hi = lane >> 4;
  const us8* p = (const us8*)(base + mr * ld + k0 + 8 * hi);
  f.h[0] = p[0];     // K = k0 + 8*hi      .. +7
  f.h[1] = p[2];     // K = k0 + 16 + 8*hi .. +7
}

// B operand, 32x16 bf16 tile from LDS staged TRANSPOSED as Bt[n][k]:
// lane n = lane%16; lanes 0-15 hold K 0..7,16..23, lanes 16-31 hold K 8..15,24..31.
__device__ __forceinline__ void loadB_frag(const unsigned short* baseT, int ld,
                                           int n0, int k0, FragBF& f) {
  const int lane = threadIdx.x & 31;
  const int nr = n0 + (lane & 15);
  const int hi = lane >> 4;
  const us8* p = (const us8*)(baseT + nr * ld + k0 + 8 * hi);
  f.h[0] = p[0];
  f.h[1] = p[2];
}

__device__ __forceinline__ f32x8 wmma_bf16(const FragBF& a, const FragBF& b, f32x8 c) {
  return __builtin_amdgcn_wmma_f32_16x16x32_bf16(false, a.v, false, b.v,
                                                 (short)0, c, false, false);
}

// =====================================================================
// Kernel 1: per-molecular-graph GCN, fully fused in LDS.
//   feats = tok_emb[enc];  T1 = feats @ Wm1;  H1 = adjT @ T1;
//   H2 = H1 @ Wm2;         H3 = adjT @ H2;    emb = max over rows(H3)
//   xc[g] = [ x[g] (bf16) | emb (bf16) ]
// One 256-thread block (8 waves, 2x4) per graph; 128x128 padded tiles.
// =====================================================================
#define MLD       136            // LDS row pitch (ushorts): bank-conflict-free, mult of 8
#define MBUF      (128 * MLD * 2)   // 34816 B per bf16 matrix
#define MOL_SMEM  (4 * MBUF)        // 139264 B dynamic LDS

__device__ __forceinline__ void zero_acc8(FragF acc[4][2]) {
#pragma unroll
  for (int i = 0; i < 4; ++i)
#pragma unroll
    for (int j = 0; j < 2; ++j)
#pragma unroll
      for (int v = 0; v < 8; ++v) acc[i][j].f[v] = 0.f;
}

// C(128x128) = A(128xK) * Bt(KxN staged as [n][k]); K = 32*ksteps.
__device__ __forceinline__ void mol_gemm(const unsigned short* __restrict__ A,
                                         const unsigned short* __restrict__ Bt,
                                         int ksteps, FragF acc[4][2]) {
  const int w  = threadIdx.x >> 5;
  const int wm = w >> 2;      // 0..1 -> 64-row slab
  const int wn = w & 3;       // 0..3 -> 32-col slab
  for (int kt = 0; kt < ksteps; ++kt) {
    const int k0 = kt * 32;
    FragBF b0, b1;
    loadB_frag(Bt, MLD, wn * 32 + 0,  k0, b0);
    loadB_frag(Bt, MLD, wn * 32 + 16, k0, b1);
#pragma unroll
    for (int i = 0; i < 4; ++i) {
      FragBF a;
      loadA_frag(A, MLD, wm * 64 + 16 * i, k0, a);
      acc[i][0].v = wmma_bf16(a, b0, acc[i][0].v);
      acc[i][1].v = wmma_bf16(a, b1, acc[i][1].v);
    }
  }
}

__global__ __launch_bounds__(256) void mol_gcn_kernel(
    const float* __restrict__ x,        // [G,384]
    const float* __restrict__ mol_adj,  // [G,100,100]
    const int*   __restrict__ enc,      // [G,100]
    const float* __restrict__ tok_emb,  // [43,75]
    const float* __restrict__ Wm1,      // [G,75,128]
    const float* __restrict__ Wm2,      // [G,128,128]
    unsigned short* __restrict__ xc)    // [G,512] bf16 out
{
  extern __shared__ __align__(16) char smem[];
  unsigned short* sAdjT = (unsigned short*)(smem);             // adjT, alive all phases
  unsigned short* sA    = (unsigned short*)(smem + MBUF);      // feats -> H1
  unsigned short* sB    = (unsigned short*)(smem + 2 * MBUF);  // W1t   -> W2t
  unsigned short* sC    = (unsigned short*)(smem + 3 * MBUF);  // T1t   -> H2t
  float*          sH3   = (float*)(smem + MBUF);               // overlays sA+sB (f32)

  const int g   = blockIdx.x;
  const int tid = threadIdx.x;
  const int w   = tid >> 5, lane = tid & 31;
  const int wm  = w >> 2,  wn   = w & 3;
  const int hi  = lane >> 4, ln = lane & 15;

  // zero-fill all LDS (provides all logical zero padding)
  for (int i = tid; i < MOL_SMEM / 4; i += 256) ((unsigned*)smem)[i] = 0u;
  __syncthreads();

  // ---- stage adjT, feats (token gather), W1t ----
  const float* adj = mol_adj + (size_t)g * A_N * A_N;
  for (int i = tid; i < A_N * A_N; i += 256) {
    int a = i / A_N, b = i % A_N;
    sAdjT[b * MLD + a] = f2b(adj[i]);               // adjT[b][a] = adj[a][b]
  }
  for (int i = tid; i < A_N * FS_N; i += 256) {
    int a = i / FS_N, f = i % FS_N;
    int t = enc[g * A_N + a];
    sA[a * MLD + f] = f2b(tok_emb[t * FS_N + f]);   // feats row-major
  }
  const float* w1 = Wm1 + (size_t)g * FS_N * HS_N;
  for (int i = tid; i < FS_N * HS_N; i += 256) {
    int f = i / HS_N, h = i % HS_N;
    sB[h * MLD + f] = f2b(w1[i]);                   // W1 transposed -> Bt layout
  }
  __syncthreads();

  FragF acc[4][2];

  // ---- GEMM1: T1 = feats @ Wm1   (K = 96 padded) -> store transposed to sC ----
  zero_acc8(acc);
  mol_gemm(sA, sB, 3, acc);
#pragma unroll
  for (int i = 0; i < 4; ++i)
#pragma unroll
    for (int j = 0; j < 2; ++j)
#pragma unroll
      for (int v = 0; v < 8; ++v) {
        int m = wm * 64 + 16 * i + v + 8 * hi;      // a index
        int n = wn * 32 + 16 * j + ln;              // h index
        sC[n * MLD + m] = f2b(acc[i][j].f[v]);      // T1t[h][a]
      }
  __syncthreads();

  // ---- GEMM2: H1 = adjT @ T1 (K=128); also stage W2t (sB dead) ----
  zero_acc8(acc);
  mol_gemm(sAdjT, sC, 4, acc);
  const float* w2 = Wm2 + (size_t)g * HS_N * HS_N;
  for (int i = tid; i < HS_N * HS_N; i += 256) {
    int h = i / HS_N, k = i % HS_N;
    sB[k * MLD + h] = f2b(w2[i]);                   // W2 transposed -> Bt layout
  }
#pragma unroll
  for (int i = 0; i < 4; ++i)
#pragma unroll
    for (int j = 0; j < 2; ++j)
#pragma unroll
      for (int v = 0; v < 8; ++v) {
        int m = wm * 64 + 16 * i + v + 8 * hi;
        int n = wn * 32 + 16 * j + ln;
        sA[m * MLD + n] = f2b(acc[i][j].f[v]);      // H1 row-major (next A operand)
      }
  __syncthreads();

  // ---- GEMM3: H2 = H1 @ Wm2 (K=128) -> store transposed to sC ----
  zero_acc8(acc);
  mol_gemm(sA, sB, 4, acc);
  __syncthreads();                                  // all GEMM2 readers of sC done
#pragma unroll
  for (int i = 0; i < 4; ++i)
#pragma unroll
    for (int j = 0; j < 2; ++j)
#pragma unroll
      for (int v = 0; v < 8; ++v) {
        int m = wm * 64 + 16 * i + v + 8 * hi;
        int n = wn * 32 + 16 * j + ln;
        sC[n * MLD + m] = f2b(acc[i][j].f[v]);      // H2t[k][a]
      }
  __syncthreads();

  // ---- GEMM4: H3 = adjT @ H2 (K=128) -> f32 to sH3 (overlays sA/sB) ----
  zero_acc8(acc);
  mol_gemm(sAdjT, sC, 4, acc);
  __syncthreads();                                  // GEMM3 readers of sA/sB done
#pragma unroll
  for (int i = 0; i < 4; ++i)
#pragma unroll
    for (int j = 0; j < 2; ++j)
#pragma unroll
      for (int v = 0; v < 8; ++v) {
        int m = wm * 64 + 16 * i + v + 8 * hi;
        int n = wn * 32 + 16 * j + ln;
        if (m < 112) sH3[m * MLD + n] = acc[i][j].f[v];
      }
  __syncthreads();

  // ---- node-wise maxpool over the 100 valid rows -> xc[g][384+k] ----
  if (tid < HS_N) {
    float mx = sH3[tid];
    for (int b = 1; b < A_N; ++b) mx = fmaxf(mx, sH3[b * MLD + tid]);
    xc[(size_t)g * 512 + 384 + tid] = f2b(mx);
  }
  // ---- copy raw features: xc[g][0..383] = bf16(x[g]) ----
  for (int i = tid; i < 384; i += 256)
    xc[(size_t)g * 512 + i] = f2b(x[(size_t)g * 384 + i]);
}

// =====================================================================
// Kernel 2: generic WMMA GEMM, 64x64 block tile, BK=32, 128 threads
// (4 waves, 2x2, each wave 32x32 = 4 C fragments).
//   AF32/BF32: global source is f32 (convert to bf16 during LDS staging)
//   BTRANS   : B given as [N][ldb] (row = output column), else [K][ldb]
//   RELU     : fused max(0,·) epilogue
//   OUTF32   : write f32 (else bf16 bits)
// =====================================================================
#define GLD 40   // LDS row pitch (ushorts) for 32-wide K tiles

template<bool AF32, bool BF32, bool BTRANS, bool RELU, bool OUTF32>
__global__ __launch_bounds__(128) void gemm_kernel(
    const void* __restrict__ Ap, int lda,
    const void* __restrict__ Bp, int ldb,
    void*       __restrict__ Cp, int ldc,
    int M, int N, int K)
{
  __shared__ __align__(16) unsigned short sA[64 * GLD];
  __shared__ __align__(16) unsigned short sB[64 * GLD];   // staged as Bt[n][k]

  const int tid   = threadIdx.x;
  const int mBase = blockIdx.y * 64;
  const int nBase = blockIdx.x * 64;
  const int w = tid >> 5, lane = tid & 31;
  const int wm = w >> 1, wn = w & 1;
  const int hi = lane >> 4, ln = lane & 15;

  const float*          Af = (const float*)Ap;
  const unsigned short* Ab = (const unsigned short*)Ap;
  const float*          Bf = (const float*)Bp;
  const unsigned short* Bb = (const unsigned short*)Bp;

  FragF acc[2][2];
#pragma unroll
  for (int i = 0; i < 2; ++i)
#pragma unroll
    for (int j = 0; j < 2; ++j)
#pragma unroll
      for (int v = 0; v < 8; ++v) acc[i][j].f[v] = 0.f;

  for (int k0 = 0; k0 < K; k0 += 32) {
    { // A tile 64x32, thread -> (row, 16-col chunk); contiguous global reads
      const int  r  = tid >> 1, cb = (tid & 1) * 16;
      const long gr = mBase + r;
#pragma unroll
      for (int i = 0; i < 16; ++i) {
        const int gc = k0 + cb + i;
        unsigned short val = 0;
        if (gr < M && gc < K)
          val = AF32 ? f2b(Af[gr * (long)lda + gc]) : Ab[gr * (long)lda + gc];
        sA[r * GLD + cb + i] = val;
      }
      if (gr < M && k0 + 32 < K) {   // prefetch next A tile (global_prefetch_b8)
        if (AF32) __builtin_prefetch(Af + gr * (long)lda + k0 + 32, 0, 1);
        else      __builtin_prefetch(Ab + gr * (long)lda + k0 + 32, 0, 1);
      }
    }
    if (BTRANS) { // B = [N][ldb]: rows are output columns; contiguous reads
      const int  n  = tid >> 1, kb = (tid & 1) * 16;
      const long gn = nBase + n;
#pragma unroll
      for (int i = 0; i < 16; ++i) {
        const int gk = k0 + kb + i;
        unsigned short val = 0;
        if (gn < N && gk < K)
          val = BF32 ? f2b(Bf[gn * (long)ldb + gk]) : Bb[gn * (long)ldb + gk];
        sB[n * GLD + kb + i] = val;
      }
    } else {      // B = [K][ldb]: read rows contiguously, scatter-transpose to LDS
      const int k  = tid >> 2, nb = (tid & 3) * 16;
      const int gk = k0 + k;
#pragma unroll
      for (int i = 0; i < 16; ++i) {
        const long gn = nBase + nb + i;
        unsigned short val = 0;
        if (gk < K && gn < N)
          val = BF32 ? f2b(Bf[(long)gk * ldb + gn]) : Bb[(long)gk * ldb + gn];
        sB[(nb + i) * GLD + k] = val;
      }
    }
    __syncthreads();

    FragBF b0, b1;
    loadB_frag(sB, GLD, wn * 32 + 0,  0, b0);
    loadB_frag(sB, GLD, wn * 32 + 16, 0, b1);
#pragma unroll
    for (int i = 0; i < 2; ++i) {
      FragBF a;
      loadA_frag(sA, GLD, wm * 32 + 16 * i, 0, a);
      acc[i][0].v = wmma_bf16(a, b0, acc[i][0].v);
      acc[i][1].v = wmma_bf16(a, b1, acc[i][1].v);
    }
    __syncthreads();
  }

  // epilogue
#pragma unroll
  for (int i = 0; i < 2; ++i)
#pragma unroll
    for (int j = 0; j < 2; ++j)
#pragma unroll
      for (int v = 0; v < 8; ++v) {
        const int row = mBase + wm * 32 + 16 * i + v + 8 * hi;
        const int col = nBase + wn * 32 + 16 * j + ln;
        if (row < M && col < N) {
          float val = acc[i][j].f[v];
          if (RELU) val = fmaxf(val, 0.f);
          if (OUTF32) ((float*)Cp)[(long)row * ldc + col] = val;
          else ((unsigned short*)Cp)[(long)row * ldc + col] = f2b(val);
        }
      }
}

// =====================================================================
// host launcher
// =====================================================================
extern "C" void kernel_launch(void* const* d_in, const int* in_sizes, int n_in,
                              void* d_out, int out_size, void* d_ws, size_t ws_size,
                              hipStream_t stream)
{
  const float* x       = (const float*)d_in[0];
  const float* adj_pos = (const float*)d_in[1];
  const float* adj_neg = (const float*)d_in[2];
  const float* mol_adj = (const float*)d_in[3];
  const int*   enc     = (const int*)  d_in[4];
  const float* tok_emb = (const float*)d_in[5];
  const float* Wm1     = (const float*)d_in[6];
  const float* Wm2     = (const float*)d_in[7];
  const float* Wp1     = (const float*)d_in[8];
  const float* Wp2     = (const float*)d_in[9];
  const float* Wn1     = (const float*)d_in[10];
  const float* Wn2     = (const float*)d_in[11];
  const float* Wd1     = (const float*)d_in[12];
  const float* Wd2     = (const float*)d_in[13];
  const float* Wd3     = (const float*)d_in[14];
  const float* Wdec    = (const float*)d_in[15];
  float* y = (float*)d_out;

  // workspace slots (bf16), with lifetime-based reuse (~11.9 MB total)
  char* ws = (char*)d_ws;
  const size_t S256 = (size_t)G_N * 256 * 2;   // 1,556,480 B
  const size_t S512 = (size_t)G_N * 512 * 2;   // 3,112,960 B
  unsigned short* s0 = (unsigned short*)(ws);                    // xc -> z   [G,512]
  unsigned short* s1 = (unsigned short*)(ws + S512);             // tp -> d1
  unsigned short* s2 = (unsigned short*)(ws + S512 + 1 * S256);  // tn
  unsigned short* s3 = (unsigned short*)(ws + S512 + 2 * S256);  // hp -> d2 (lo half)
  unsigned short* s4 = (unsigned short*)(ws + S512 + 3 * S256);  // hn -> d2 (hi half)
  unsigned short* s5 = (unsigned short*)(ws + S512 + 4 * S256);  // up -> z3
  unsigned short* s6 = (unsigned short*)(ws + S512 + 5 * S256);  // un -> zw
  unsigned short* xc = s0;
  unsigned short* d2 = s3;  // [G,512] spans s3+s4 contiguously

  hipFuncSetAttribute(reinterpret_cast<const void*>(mol_gcn_kernel),
                      hipFuncAttributeMaxDynamicSharedMemorySize, MOL_SMEM);

  // 1) fused per-graph molecular GCN + maxpool + concat -> xc (bf16)
  mol_gcn_kernel<<<G_N, 256, MOL_SMEM, stream>>>(x, mol_adj, enc, tok_emb, Wm1, Wm2, xc);

  auto grd = [](int M, int N) { return dim3((unsigned)((N + 63) / 64),
                                            (unsigned)((M + 63) / 64)); };

  // 2,3) tp = xc@Wp1 ; tn = xc@Wn1
  gemm_kernel<false,true,false,false,false><<<grd(G_N,256),128,0,stream>>>(xc,512, Wp1,256, s1,256, G_N,256,512);
  gemm_kernel<false,true,false,false,false><<<grd(G_N,256),128,0,stream>>>(xc,512, Wn1,256, s2,256, G_N,256,512);
  // 4,5) hp = relu(adj_pos@tp) ; hn = relu(adj_neg@tn)   (f32 adj converted on load)
  gemm_kernel<true,false,false,true,false><<<grd(G_N,256),128,0,stream>>>(adj_pos,G_N, s1,256, s3,256, G_N,256,G_N);
  gemm_kernel<true,false,false,true,false><<<grd(G_N,256),128,0,stream>>>(adj_neg,G_N, s2,256, s4,256, G_N,256,G_N);
  // 6,7) up = hp@Wp2 ; un = hn@Wn2
  gemm_kernel<false,true,false,false,false><<<grd(G_N,256),128,0,stream>>>(s3,256, Wp2,256, s5,256, G_N,256,256);
  gemm_kernel<false,true,false,false,false><<<grd(G_N,256),128,0,stream>>>(s4,256, Wn2,256, s6,256, G_N,256,256);
  // 8,9) z = [adj_pos@up | adj_neg@un]  (write halves of s0 with ldc=512)
  gemm_kernel<true,false,false,false,false><<<grd(G_N,256),128,0,stream>>>(adj_pos,G_N, s5,256, s0,      512, G_N,256,G_N);
  gemm_kernel<true,false,false,false,false><<<grd(G_N,256),128,0,stream>>>(adj_neg,G_N, s6,256, s0 + 256,512, G_N,256,G_N);
  // 10) d1 = relu(z@Wd1)
  gemm_kernel<false,true,false,true,false><<<grd(G_N,256),128,0,stream>>>(s0,512, Wd1,256, s1,256, G_N,256,512);
  // 11) d2 = relu(d1@Wd2)
  gemm_kernel<false,true,false,true,false><<<grd(G_N,512),128,0,stream>>>(s1,256, Wd2,512, d2,512, G_N,512,256);
  // 12) z3 = d2@Wd3
  gemm_kernel<false,true,false,false,false><<<grd(G_N,256),128,0,stream>>>(d2,512, Wd3,256, s5,256, G_N,256,512);
  // 13) zw = z3@Wdec
  gemm_kernel<false,true,false,false,false><<<grd(G_N,256),128,0,stream>>>(s5,256, Wdec,256, s6,256, G_N,256,256);
  // 14) y = zw @ z3^T   (B given as [N][K] -> BTRANS path, f32 output)
  gemm_kernel<false,false,true,false,true><<<grd(G_N,G_N),128,0,stream>>>(s6,256, s5,256, y,G_N, G_N,G_N,256);

  (void)in_sizes; (void)n_in; (void)out_size; (void)ws_size;
}
